// AttentionBlock_29291676959393
// MI455X (gfx1250) — matrix operand order repaired and checked by reading, not compiled, not run on hardware
//
#include <hip/hip_runtime.h>

#define DEV __device__ __forceinline__

typedef __attribute__((ext_vector_type(16))) _Float16 v16h;
typedef __attribute__((ext_vector_type(8)))  _Float16 v8h;
typedef __attribute__((ext_vector_type(8)))  float    v8f;

static constexpr int N  = 4;
static constexpr int L  = 2048;
static constexpr int D  = 1024;
static constexpr int DC = 512;
static constexpr int NH = 16;
static constexpr int E  = 64;
static constexpr float EPS = 1e-6f;

DEV v8f wmma32(v16h a, v16h b, v8f c) {
  return __builtin_amdgcn_wmma_f32_16x16x32_f16(false, a, false, b, (short)0, c,
                                                false, false);
}

// A-matrix 16x32 (MxK), row-major source, K contiguous.
// ISA layout: lane m (0-15): K {0..7,16..23}; lane m+16: K {8..15,24..31}.
DEV v16h load_a16x32(const _Float16* base, int ld) {
  int lane = threadIdx.x & 31;
  int m = lane & 15, h = lane >> 4;
  const _Float16* p = base + m * ld + h * 8;
  v8h lo = *(const v8h*)p;
  v8h hi = *(const v8h*)(p + 16);
  v16h r;
#pragma unroll
  for (int i = 0; i < 8; ++i) { r[i] = lo[i]; r[i + 8] = hi[i]; }
  return r;
}

// B-matrix 32x16 (KxN) given as B^T rows: element (k,n) at base[n*ld + k].
// ISA layout: lane n: K {0..7,16..23}; lane n+16: K {8..15,24..31}.
DEV v16h load_bt16x32(const _Float16* base, int ld) {
  int lane = threadIdx.x & 31;
  int n = lane & 15, h = lane >> 4;
  const _Float16* p = base + n * ld + h * 8;
  v8h lo = *(const v8h*)p;
  v8h hi = *(const v8h*)(p + 16);
  v16h r;
#pragma unroll
  for (int i = 0; i < 8; ++i) { r[i] = lo[i]; r[i + 8] = hi[i]; }
  return r;
}

// Load one 32x64-output K-step operand set: 2 A subtiles + 4 B subtiles.
DEV void load_set(const _Float16* A, const _Float16* B, int kk,
                  v16h a[2], v16h b[4]) {
  a[0] = load_a16x32(A + kk, D);
  a[1] = load_a16x32(A + (size_t)16 * D + kk, D);
#pragma unroll
  for (int j = 0; j < 4; ++j)
    b[j] = load_bt16x32(B + (size_t)(j * 16) * D + kk, D);
}

DEV void mma_step(v8f acc[8], const v16h a[2], const v16h b[4]) {
#pragma unroll
  for (int ms = 0; ms < 2; ++ms)
#pragma unroll
    for (int j = 0; j < 4; ++j)
      acc[ms * 4 + j] = wmma32(a[ms], b[j], acc[ms * 4 + j]);
}

// ---------------- elementwise / prep kernels ----------------

__global__ void k_cvt_f16(const float* __restrict__ src,
                          _Float16* __restrict__ dst, int n) {
  int i = blockIdx.x * blockDim.x + threadIdx.x;
  if (i < n) dst[i] = (_Float16)src[i];
}

__global__ void k_gain(const float* __restrict__ cond,
                       const float* __restrict__ norm_w,
                       float* __restrict__ gain) {
  int i = blockIdx.x * blockDim.x + threadIdx.x;  // [0, N*D)
  int ni = i / D, dd = i % D;
  const float* c = cond + ni * DC;
  const float* w = norm_w + (size_t)dd * DC;
  float s = 0.f;
  for (int j = 0; j < DC; ++j) s += c[j] * w[j];
  gain[i] = s + 1.0f;
}

__global__ void k_rmsnorm(const float* __restrict__ x,
                          const float* __restrict__ gain,
                          _Float16* __restrict__ xn) {
  int row = blockIdx.x;  // N*L rows
  int ni = row / L;
  const float* xr = x + (size_t)row * D;
  int tid = threadIdx.x;  // 256
  float ss = 0.f;
#pragma unroll
  for (int i = tid; i < D; i += 256) { float v = xr[i]; ss += v * v; }
  __shared__ float red[256];
  red[tid] = ss;
  __syncthreads();
  for (int s = 128; s > 0; s >>= 1) {
    if (tid < s) red[tid] += red[tid + s];
    __syncthreads();
  }
  float inv = rsqrtf(red[0] * (1.0f / D) + EPS);
  const float* g = gain + ni * D;
  _Float16* o = xn + (size_t)row * D;
#pragma unroll
  for (int i = tid; i < D; i += 256) o[i] = (_Float16)(xr[i] * g[i] * inv);
}

// ---------------- QKV GEMM + q/k cosine norm + V transpose ----------------
// wave => (32-row tile rt, t in {q,k,v}, head h): 32 rows x 64 cols.
// Software-pipelined: loads for K-step i+1 issue before WMMAs of step i.
__global__ void __launch_bounds__(128) k_qkv(
    const _Float16* __restrict__ xn, const _Float16* __restrict__ w16,
    const float* __restrict__ head_scale, _Float16* __restrict__ q,
    _Float16* __restrict__ k, _Float16* __restrict__ vT) {
  int w = blockIdx.x * 4 + (threadIdx.x >> 5);
  int lane = threadIdx.x & 31;
  int rt = w & 255;   // 256 row tiles of 32 over N*L=8192
  int th = w >> 8;    // 0..47
  int t = th >> 4, h = th & 15;
  const _Float16* A = xn + (size_t)rt * 32 * D;
  const _Float16* B = w16 + (size_t)(t * D + h * E) * D;
  v8f acc[8] = {};
  v16h a0[2], b0[4], a1[2], b1[4];
  load_set(A, B, 0, a0, b0);
  for (int kk = 0; kk < D; kk += 64) {
    load_set(A, B, kk + 32, a1, b1);
    mma_step(acc, a0, b0);
    int k2 = (kk + 64 < D) ? kk + 64 : 0;  // tail reload is dead but harmless
    load_set(A, B, k2, a0, b0);
    mma_step(acc, a1, b1);
  }

  int colL = lane & 15, rowb = (lane >> 4) * 8;
#pragma unroll
  for (int ms = 0; ms < 2; ++ms) {
    const v8f* ac = &acc[ms * 4];
    int r0 = rt * 32 + ms * 16;
    int ni = r0 / L;  // tile never crosses n boundary (L%16==0)
    if (t == 2) {
      // V: store transposed [n, nh, e, L]
#pragma unroll
      for (int j4 = 0; j4 < 4; ++j4)
#pragma unroll
        for (int j = 0; j < 8; ++j) {
          int e = j4 * 16 + colL;
          int li = (r0 + rowb + j) % L;
          vT[((size_t)(ni * NH + h) * E + e) * L + li] = (_Float16)ac[j4][j];
        }
    } else {
      float sc = sqrtf(head_scale[h]);
      _Float16* dst = (t == 0) ? q : k;
#pragma unroll
      for (int j = 0; j < 8; ++j) {
        float s = 0.f;
#pragma unroll
        for (int j4 = 0; j4 < 4; ++j4) s += ac[j4][j] * ac[j4][j];
#pragma unroll
        for (int off = 8; off >= 1; off >>= 1) s += __shfl_xor(s, off, 32);
        float mul = sc * rsqrtf(s + EPS);
        int li = (r0 + rowb + j) % L;
        size_t base = ((size_t)(ni * NH + h) * L + li) * E;
#pragma unroll
        for (int j4 = 0; j4 < 4; ++j4)
          dst[base + j4 * 16 + colL] = (_Float16)(ac[j4][j] * mul);
      }
    }
  }
}

// ---------------- flash attention ----------------
// wave => one (n,h) pair + 16-row q tile; kv in chunks of 32.
// K tiles double-buffered across chunks; V tiles loaded before softmax so
// they overlap the VALU-heavy exp/reduce phase.
__global__ void __launch_bounds__(128) k_attn(const _Float16* __restrict__ q,
                                              const _Float16* __restrict__ k,
                                              const _Float16* __restrict__ vT,
                                              _Float16* __restrict__ o) {
  __shared__ __align__(16) _Float16 Pl[4][16 * 32];
  int wid = threadIdx.x >> 5;
  int w = blockIdx.x * 4 + wid;
  int lane = threadIdx.x & 31;
  int nh = w >> 7;   // 0..63
  int qt = w & 127;  // 128 q tiles
  int ni = nh >> 4, h = nh & 15;
  const _Float16* qb = q + ((size_t)nh * L + qt * 16) * E;
  const _Float16* kb = k + (size_t)nh * L * E;
  const _Float16* vb = vT + (size_t)nh * E * L;

  v16h aq0 = load_a16x32(qb, E);       // K = e 0..31
  v16h aq1 = load_a16x32(qb + 32, E);  // K = e 32..63

  v8f O[4] = {};
  float mr[8], lr[8];
#pragma unroll
  for (int j = 0; j < 8; ++j) { mr[j] = -3.0e38f; lr[j] = 0.f; }

  int colL = lane & 15, rowb = (lane >> 4) * 8;

  // prologue: K tiles for chunk 0
  v16h bk[4];
  bk[0] = load_bt16x32(kb, E);
  bk[1] = load_bt16x32(kb + 32, E);
  bk[2] = load_bt16x32(kb + (size_t)16 * E, E);
  bk[3] = load_bt16x32(kb + (size_t)16 * E + 32, E);

  for (int c = 0; c < L; c += 32) {
    v8f s0 = {}, s1 = {};
    s0 = wmma32(aq0, bk[0], s0);
    s0 = wmma32(aq1, bk[1], s0);
    s1 = wmma32(aq0, bk[2], s1);
    s1 = wmma32(aq1, bk[3], s1);

    // prefetch next chunk's K tiles + this chunk's V tiles (overlap softmax)
    int cn = (c + 32 < L) ? c + 32 : c;
    v16h bkn[4];
    bkn[0] = load_bt16x32(kb + (size_t)cn * E, E);
    bkn[1] = load_bt16x32(kb + (size_t)cn * E + 32, E);
    bkn[2] = load_bt16x32(kb + (size_t)(cn + 16) * E, E);
    bkn[3] = load_bt16x32(kb + (size_t)(cn + 16) * E + 32, E);
    v16h vt[4];
#pragma unroll
    for (int t = 0; t < 4; ++t)
      vt[t] = load_bt16x32(vb + (size_t)(t * 16) * L + c, L);

    float alpha[8];
#pragma unroll
    for (int j = 0; j < 8; ++j) {
      float cm = fmaxf(s0[j], s1[j]);
#pragma unroll
      for (int off = 8; off >= 1; off >>= 1)
        cm = fmaxf(cm, __shfl_xor(cm, off, 32));
      float mnew = fmaxf(mr[j], cm);
      alpha[j] = __expf(mr[j] - mnew);
      float p0 = __expf(s0[j] - mnew);
      float p1 = __expf(s1[j] - mnew);
      s0[j] = p0;
      s1[j] = p1;
      float ps = p0 + p1;
#pragma unroll
      for (int off = 8; off >= 1; off >>= 1) ps += __shfl_xor(ps, off, 32);
      lr[j] = lr[j] * alpha[j] + ps;
      mr[j] = mnew;
    }
#pragma unroll
    for (int t = 0; t < 4; ++t)
#pragma unroll
      for (int j = 0; j < 8; ++j) O[t][j] *= alpha[j];

    // stage P (C layout -> LDS row-major 16x32)
#pragma unroll
    for (int j = 0; j < 8; ++j) {
      Pl[wid][(rowb + j) * 32 + colL] = (_Float16)s0[j];
      Pl[wid][(rowb + j) * 32 + 16 + colL] = (_Float16)s1[j];
    }
    __syncthreads();
    v16h pa;
    {
      int m = lane & 15, hh = lane >> 4;
#pragma unroll
      for (int i = 0; i < 8; ++i) {
        pa[i] = Pl[wid][m * 32 + hh * 8 + i];
        pa[i + 8] = Pl[wid][m * 32 + hh * 8 + 16 + i];
      }
    }
    __syncthreads();
#pragma unroll
    for (int t = 0; t < 4; ++t) O[t] = wmma32(pa, vt[t], O[t]);
#pragma unroll
    for (int t = 0; t < 4; ++t) bk[t] = bkn[t];
  }

  // normalize and store to o[n, l, D] at column block h*64
#pragma unroll
  for (int j = 0; j < 8; ++j) {
    float inv = 1.0f / lr[j];
    int li = qt * 16 + rowb + j;
    size_t base = ((size_t)ni * L + li) * D + h * E;
#pragma unroll
    for (int t = 0; t < 4; ++t)
      o[base + t * 16 + colL] = (_Float16)(O[t][j] * inv);
  }
}

// ---------------- out projection + skip ----------------
__global__ void __launch_bounds__(128) k_outproj(
    const _Float16* __restrict__ o16, const _Float16* __restrict__ w16,
    const float* __restrict__ x, float* __restrict__ out) {
  int w = blockIdx.x * 4 + (threadIdx.x >> 5);
  int lane = threadIdx.x & 31;
  int rt = w & 255;  // 32-row tile
  int ct = w >> 8;   // 0..15, 64 cols each
  const _Float16* A = o16 + (size_t)rt * 32 * D;
  const _Float16* B = w16 + (size_t)(ct * 64) * D;
  v8f acc[8] = {};
  v16h a0[2], b0[4], a1[2], b1[4];
  load_set(A, B, 0, a0, b0);
  for (int kk = 0; kk < D; kk += 64) {
    load_set(A, B, kk + 32, a1, b1);
    mma_step(acc, a0, b0);
    int k2 = (kk + 64 < D) ? kk + 64 : 0;
    load_set(A, B, k2, a0, b0);
    mma_step(acc, a1, b1);
  }
  int colL = lane & 15, rowb = (lane >> 4) * 8;
#pragma unroll
  for (int ms = 0; ms < 2; ++ms) {
    const v8f* ac = &acc[ms * 4];
#pragma unroll
    for (int j = 0; j < 8; ++j) {
      size_t row = (size_t)rt * 32 + ms * 16 + rowb + j;
#pragma unroll
      for (int t = 0; t < 4; ++t) {
        size_t idx = row * D + ct * 64 + t * 16 + colL;
        out[idx] = ac[t][j] + x[idx];
      }
    }
  }
}

extern "C" void kernel_launch(void* const* d_in, const int* in_sizes, int n_in,
                              void* d_out, int out_size, void* d_ws,
                              size_t ws_size, hipStream_t stream) {
  (void)in_sizes; (void)n_in; (void)out_size; (void)ws_size;
  const float* x      = (const float*)d_in[0];
  // d_in[1] = pos (unused: rope_cls=None)
  const float* cond   = (const float*)d_in[2];
  const float* norm_w = (const float*)d_in[3];
  const float* qkv_w  = (const float*)d_in[4];
  const float* hscale = (const float*)d_in[5];
  const float* out_w  = (const float*)d_in[6];
  float* out = (float*)d_out;

  char* ws = (char*)d_ws;
  const size_t MB = (size_t)1 << 20;
  _Float16* xn_o   = (_Float16*)(ws + 0);        // 16MB: xn, later reused as o
  _Float16* qkvw16 = (_Float16*)(ws + 16 * MB);  // 6MB
  _Float16* outw16 = (_Float16*)(ws + 22 * MB);  // 2MB
  _Float16* qbuf   = (_Float16*)(ws + 24 * MB);  // 16MB [n,nh,L,E]
  _Float16* kbuf   = (_Float16*)(ws + 40 * MB);  // 16MB [n,nh,L,E]
  _Float16* vTbuf  = (_Float16*)(ws + 56 * MB);  // 16MB [n,nh,E,L]
  float*    gain   = (float*)(ws + 72 * MB);     // 16KB

  k_cvt_f16<<<(3 * D * D + 255) / 256, 256, 0, stream>>>(qkv_w, qkvw16, 3 * D * D);
  k_cvt_f16<<<(D * D + 255) / 256, 256, 0, stream>>>(out_w, outw16, D * D);
  k_gain<<<(N * D) / 256, 256, 0, stream>>>(cond, norm_w, gain);
  k_rmsnorm<<<N * L, 256, 0, stream>>>(x, gain, xn_o);
  k_qkv<<<(256 * 48) / 4, 128, 0, stream>>>(xn_o, qkvw16, hscale, qbuf, kbuf, vTbuf);
  k_attn<<<(64 * 128) / 4, 128, 0, stream>>>(qbuf, kbuf, vTbuf, xn_o);
  k_outproj<<<(256 * 16) / 4, 128, 0, stream>>>(xn_o, outw16, x, out);
}
